// HyperCritic_88132728914137
// MI455X (gfx1250) — matrix operand order repaired and checked by reading, not compiled, not run on hardware
//
#include <hip/hip_runtime.h>

// ---------------------------------------------------------------------------
// HyperCritic forward for MI455X (gfx1250, wave32, WMMA).
//   out[b,o] = sum_{k<=512} z'[b,k] * sum_i h[b,i] * wwt[o, k*fi_pad + i]
// via v_wmma_f32_16x16x32_bf16 with A = (z outer h) built on the fly from
// LDS-staged tiles. Weights pre-converted to bf16 (fits in 192MB L2).
// Biases folded in as the k==512 slot against z'[:,512] = 1.
// Hyper kernel: 64-row batch tile (4 M-subtiles, 4 accumulators) so each
// B fragment feeds 4 WMMAs -> 4x less L2 weight traffic. ~90 MB workspace.
// ---------------------------------------------------------------------------

typedef __attribute__((ext_vector_type(16))) __bf16 v16bf;
typedef __attribute__((ext_vector_type(8)))  float  v8f;
typedef __attribute__((ext_vector_type(4)))  float  f32x4;

union Frag16 { v16bf v; f32x4 q[2]; unsigned int u[8]; };

static constexpr int BATCH   = 8192;
static constexpr int OBS     = 128;
static constexpr int ACT     = 64;
static constexpr int RWD     = 4;
static constexpr int TIN_P   = 224;               // 196 padded to 32
static constexpr int ZDIM    = 512;
static constexpr int ZLD     = 544;               // 512 + 1 (ones) + pad
static constexpr int E1      = 256;
static constexpr int E1LD    = 288;               // 256 + 1 + pad
static constexpr int AB0LD   = 160;               // 132 + 1 + pad
static constexpr int KW      = 513;               // 512 z slots + wb slot

#if __has_builtin(__builtin_amdgcn_global_load_async_to_lds_b32) && \
    __has_builtin(__builtin_amdgcn_s_wait_asynccnt)
#define USE_ASYNC_LDS 1
#endif

// ------------------------------ prep kernels -------------------------------

__global__ void prep_inputs(const float* __restrict__ obs,
                            const float* __restrict__ act,
                            const float* __restrict__ prefs,
                            __bf16* __restrict__ ab0,   // [B, 160]
                            __bf16* __restrict__ hb0) { // [B, 224]
  size_t tid = (size_t)blockIdx.x * blockDim.x + threadIdx.x;
  size_t stride = (size_t)gridDim.x * blockDim.x;
  const size_t nA = (size_t)BATCH * AB0LD;
  for (size_t i = tid; i < nA; i += stride) {
    int b = (int)(i / AB0LD), c = (int)(i % AB0LD);
    float v;
    if      (c < 128) v = obs[(size_t)b * 128 + c];
    else if (c < 132) v = prefs[(size_t)b * 4 + (c - 128)];
    else if (c == 132) v = 1.0f;
    else v = 0.0f;
    ab0[i] = (__bf16)v;
  }
  const size_t nH = (size_t)BATCH * TIN_P;
  for (size_t i = tid; i < nH; i += stride) {
    int b = (int)(i / TIN_P), c = (int)(i % TIN_P);
    float v;
    if      (c < 128) v = obs[(size_t)b * 128 + c];
    else if (c < 192) v = act[(size_t)b * 64 + (c - 128)];
    else if (c < 196) v = prefs[(size_t)b * 4 + (c - 192)];
    else v = 0.0f;
    hb0[i] = (__bf16)v;
  }
}

// set column c1 to 1.0 and columns (c1, ld) to 0 for all BATCH rows
__global__ void init_tail(__bf16* __restrict__ buf, int ld, int c1) {
  int w = ld - c1;
  size_t total = (size_t)BATCH * w;
  size_t stride = (size_t)gridDim.x * blockDim.x;
  for (size_t i = (size_t)blockIdx.x * blockDim.x + threadIdx.x; i < total; i += stride) {
    int b = (int)(i / w), c = c1 + (int)(i % w);
    buf[(size_t)b * ld + c] = (c == c1) ? (__bf16)1.0f : (__bf16)0.0f;
  }
}

// W[K,N] fp32 (+bias[N] at slot k==K) -> Bt[N,Kpad] bf16
__global__ void cvt_bt(const float* __restrict__ W, const float* __restrict__ bias,
                       __bf16* __restrict__ dst, int K, int N, int Kpad) {
  size_t total = (size_t)N * Kpad;
  size_t stride = (size_t)gridDim.x * blockDim.x;
  for (size_t i = (size_t)blockIdx.x * blockDim.x + threadIdx.x; i < total; i += stride) {
    int n = (int)(i / Kpad), k = (int)(i % Kpad);
    float v = 0.0f;
    if (k < K)       v = W[(size_t)k * N + n];
    else if (k == K) v = bias[n];
    dst[i] = (__bf16)v;
  }
}

// ww[512, fo*fi] fp32 + wb[fo*fi] fp32 -> wwt[fo, 513*fip] bf16
__global__ void cvt_ww(const float* __restrict__ ww, const float* __restrict__ wb,
                       __bf16* __restrict__ dst, int fo, int fi, int fip) {
  size_t total = (size_t)fo * KW * fip;
  size_t stride = (size_t)gridDim.x * blockDim.x;
  for (size_t idx = (size_t)blockIdx.x * blockDim.x + threadIdx.x; idx < total; idx += stride) {
    int i = (int)(idx % fip);
    size_t t = idx / fip;
    int k = (int)(t % KW);
    int o = (int)(t / KW);
    float v = 0.0f;
    if (i < fi)
      v = (k < 512) ? ww[(size_t)k * fo * fi + (size_t)o * fi + i]
                    : wb[(size_t)o * fi + i];
    dst[idx] = (__bf16)v;
  }
}

// ------------------------------ generic GEMM -------------------------------
// C[M,N] = act(A[M,Kpad] @ Bt[N,Kpad]^T). 8 waves, tile 128(M) x 64(N).
__global__ void __launch_bounds__(256) gemm_bt(
    const __bf16* __restrict__ A, int lda,
    const __bf16* __restrict__ Bt, int ldb, int Kpad, int N,
    __bf16* __restrict__ outb, int lddb,
    float*  __restrict__ outf, int lddf, int relu) {
  const int lane = threadIdx.x & 31;
  const int wv   = threadIdx.x >> 5;
  const int half = lane >> 4;
  const int l16  = lane & 15;
  const int m0 = blockIdx.y * 128 + wv * 16;
  const int n0 = blockIdx.x * 64;

  v8f acc0 = {}, acc1 = {}, acc2 = {}, acc3 = {};
  const __bf16* aRow = A + (size_t)(m0 + l16) * lda + half * 8;

  for (int kk = 0; kk < Kpad; kk += 32) {
    Frag16 af;
    af.q[0] = *(const f32x4*)(aRow + kk);        // K: half*8 .. +7
    af.q[1] = *(const f32x4*)(aRow + kk + 16);   // K: 16+half*8 .. +7
#pragma unroll
    for (int j = 0; j < 4; ++j) {
      int br = n0 + j * 16 + l16; if (br >= N) br = N - 1;
      const __bf16* bp = Bt + (size_t)br * ldb + kk + half * 8;
      Frag16 bf;
      bf.q[0] = *(const f32x4*)bp;               // K: half*8 .. +7
      bf.q[1] = *(const f32x4*)(bp + 16);        // K: 16+half*8 .. +7
      v8f& acc = (j == 0) ? acc0 : (j == 1) ? acc1 : (j == 2) ? acc2 : acc3;
      acc = __builtin_amdgcn_wmma_f32_16x16x32_bf16(
          false, af.v, false, bf.v, (short)0, acc, false, false);
    }
  }
#pragma unroll
  for (int j = 0; j < 4; ++j) {
    int n = n0 + j * 16 + l16;
    if (n >= N) continue;
    v8f acc = (j == 0) ? acc0 : (j == 1) ? acc1 : (j == 2) ? acc2 : acc3;
#pragma unroll
    for (int r = 0; r < 8; ++r) {
      int row = m0 + half * 8 + r;
      float v = acc[r];
      if (relu) v = fmaxf(v, 0.0f);
      if (outb) outb[(size_t)row * lddb + n] = (__bf16)v;
      if (outf) outf[(size_t)row * lddf + n] = v;
    }
  }
}

// --------------------------- hypernet contraction --------------------------
// Tile: MROWS = 16*MSUB batch rows per block; each wave owns one 16-col
// output tile and MSUB accumulators; one B fragment feeds MSUB WMMAs.
// h and z tiles staged to LDS (async-to-LDS when toolchain exposes it).
template <int NC, int MSUB>  // NC = fi_pad/32
__global__ void hyper_layer(
    const __bf16* __restrict__ hb,            // [B, FIP]
    const __bf16* __restrict__ zb,            // [B, 544]
    const __bf16* __restrict__ wwt,           // [fo, 513*FIP]
    int fo, float* __restrict__ out, int ldo) {
  constexpr int FIP = NC * 32;
  constexpr int MROWS = 16 * MSUB;
  __shared__ __bf16 zt[MROWS * ZLD];
  __shared__ __bf16 ht[MROWS * FIP];
  const int lane = threadIdx.x & 31;
  const int wv   = threadIdx.x >> 5;
  const int half = lane >> 4;
  const int l16  = lane & 15;
  const int b0 = blockIdx.x * MROWS;
  const int n0 = (blockIdx.y * (blockDim.x >> 5) + wv) * 16;

  { // stage z and h tiles (contiguous rows) to LDS
    const unsigned int* zsrc = (const unsigned int*)(zb + (size_t)b0 * ZLD);
    const unsigned int* hsrc = (const unsigned int*)(hb + (size_t)b0 * FIP);
    unsigned int* zdst = (unsigned int*)zt;
    unsigned int* hdst = (unsigned int*)ht;
#ifdef USE_ASYNC_LDS
    for (int t = threadIdx.x; t < MROWS * ZLD / 2; t += blockDim.x)
      __builtin_amdgcn_global_load_async_to_lds_b32(
          (int*)&zsrc[t], (int*)&zdst[t], 0, 0);
    for (int t = threadIdx.x; t < MROWS * FIP / 2; t += blockDim.x)
      __builtin_amdgcn_global_load_async_to_lds_b32(
          (int*)&hsrc[t], (int*)&hdst[t], 0, 0);
    __builtin_amdgcn_s_wait_asynccnt(0);
#else
    for (int t = threadIdx.x; t < MROWS * ZLD / 2; t += blockDim.x) zdst[t] = zsrc[t];
    for (int t = threadIdx.x; t < MROWS * FIP / 2; t += blockDim.x) hdst[t] = hsrc[t];
#endif
  }
  __syncthreads();

  int wrow = n0 + l16; if (wrow >= fo) wrow = fo - 1;
  const __bf16* wbase = wwt + (size_t)wrow * ((size_t)KW * FIP) + half * 8;

  v8f acc[MSUB];
#pragma unroll
  for (int s = 0; s < MSUB; ++s) acc[s] = (v8f){};

  Frag16 hf[MSUB];
#pragma unroll 1
  for (int c = 0; c < NC; ++c) {
    // per-chunk h fragments for each M-subtile (A-layout K halves)
#pragma unroll
    for (int s = 0; s < MSUB; ++s) {
      const __bf16* hp = ht + (size_t)(s * 16 + l16) * FIP + c * 32 + half * 8;
      hf[s].q[0] = *(const f32x4*)hp;
      hf[s].q[1] = *(const f32x4*)(hp + 16);
    }
    const __bf16* wp = wbase + c * 32;
#pragma unroll 2
    for (int k = 0; k < KW; ++k) {
      Frag16 bf;                                  // B-layout K halves
      bf.q[0] = *(const f32x4*)wp;
      bf.q[1] = *(const f32x4*)(wp + 16);
      __builtin_prefetch(wp + FIP, 0, 0);         // next k's weight row
#pragma unroll
      for (int s = 0; s < MSUB; ++s) {
        unsigned short zk = *(const unsigned short*)&zt[(s * 16 + l16) * ZLD + k];
        unsigned int zz = ((unsigned int)zk << 16) | zk;
        Frag16 zs;
#pragma unroll
        for (int e = 0; e < 8; ++e) zs.u[e] = zz;
        v16bf a = hf[s].v * zs.v;                 // packed bf16 outer product
        acc[s] = __builtin_amdgcn_wmma_f32_16x16x32_bf16(
            false, a, false, bf.v, (short)0, acc[s], false, false);
      }
      wp += FIP;
    }
  }

  int n = n0 + l16;
  if (n < fo) {
#pragma unroll
    for (int s = 0; s < MSUB; ++s)
#pragma unroll
      for (int r = 0; r < 8; ++r)
        out[(size_t)(b0 + s * 16 + half * 8 + r) * ldo + n] = acc[s][r];
  }
}

// ------------------------------- epilogue ----------------------------------
__global__ void combine_layer(const float* __restrict__ mv,
                              const float* __restrict__ bg,
                              const float* __restrict__ sg, int fo,
                              __bf16* __restrict__ hnext,
                              float* __restrict__ fout, int relu) {
  size_t total = (size_t)BATCH * fo;
  size_t stride = (size_t)gridDim.x * blockDim.x;
  for (size_t i = (size_t)blockIdx.x * blockDim.x + threadIdx.x; i < total; i += stride) {
    float v = sg[i] * (mv[i] + bg[i]);
    if (relu) v = fmaxf(v, 0.0f);
    if (hnext) hnext[i] = (__bf16)v;
    if (fout)  fout[i] = v;
  }
}

// ------------------------------- launcher ----------------------------------
static inline size_t alignUp(size_t x) { return (x + 255) & ~(size_t)255; }

extern "C" void kernel_launch(void* const* d_in, const int* in_sizes, int n_in,
                              void* d_out, int out_size, void* d_ws, size_t ws_size,
                              hipStream_t stream) {
  (void)in_sizes; (void)n_in; (void)out_size; (void)ws_size;
  const float* obs    = (const float*)d_in[0];
  const float* act    = (const float*)d_in[1];
  const float* prefs  = (const float*)d_in[2];
  const float* emb_w1 = (const float*)d_in[3];
  const float* emb_b1 = (const float*)d_in[4];
  const float* emb_w2 = (const float*)d_in[5];
  const float* emb_b2 = (const float*)d_in[6];
  const float* L[3][6];   // ww, wb, bw, bb, sw, sb
  for (int l = 0; l < 3; ++l)
    for (int j = 0; j < 6; ++j) L[l][j] = (const float*)d_in[7 + 6 * l + j];
  const int FO[3]  = {128, 128, 4};
  const int FI[3]  = {196, 128, 128};
  const int FIP[3] = {224, 128, 128};

  char* p = (char*)d_ws;
  auto take = [&](size_t bytes) { void* r = (void*)p; p += alignUp(bytes); return r; };
  __bf16* ab0  = (__bf16*)take((size_t)BATCH * AB0LD * 2);
  __bf16* hb0  = (__bf16*)take((size_t)BATCH * TIN_P * 2);
  __bf16* hb1  = (__bf16*)take((size_t)BATCH * 128 * 2);
  __bf16* hb2  = (__bf16*)take((size_t)BATCH * 128 * 2);
  __bf16* zb1  = (__bf16*)take((size_t)BATCH * E1LD * 2);
  __bf16* zb   = (__bf16*)take((size_t)BATCH * ZLD * 2);
  __bf16* btE1 = (__bf16*)take((size_t)E1 * AB0LD * 2);
  __bf16* btE2 = (__bf16*)take((size_t)ZDIM * E1LD * 2);
  __bf16* bwt[3], *swt[3], *wwt[3];
  for (int l = 0; l < 3; ++l) {
    bwt[l] = (__bf16*)take((size_t)FO[l] * ZLD * 2);
    swt[l] = (__bf16*)take((size_t)FO[l] * ZLD * 2);
    wwt[l] = (__bf16*)take((size_t)FO[l] * KW * FIP[l] * 2);
  }
  float* mv = (float*)take((size_t)BATCH * 128 * 4);
  float* bg = (float*)take((size_t)BATCH * 128 * 4);
  float* sg = (float*)take((size_t)BATCH * 128 * 4);

  // --- prep / conversion ---
  prep_inputs<<<1024, 256, 0, stream>>>(obs, act, prefs, ab0, hb0);
  init_tail<<<512, 256, 0, stream>>>(zb1, E1LD, 256);
  init_tail<<<512, 256, 0, stream>>>(zb, ZLD, 512);
  cvt_bt<<<512, 256, 0, stream>>>(emb_w1, emb_b1, btE1, 132, E1, AB0LD);
  cvt_bt<<<512, 256, 0, stream>>>(emb_w2, emb_b2, btE2, 256, ZDIM, E1LD);
  for (int l = 0; l < 3; ++l) {
    cvt_bt<<<512, 256, 0, stream>>>(L[l][2], L[l][3], bwt[l], ZDIM, FO[l], ZLD);
    cvt_bt<<<512, 256, 0, stream>>>(L[l][4], L[l][5], swt[l], ZDIM, FO[l], ZLD);
    cvt_ww<<<2048, 256, 0, stream>>>(L[l][0], L[l][1], wwt[l], FO[l], FI[l], FIP[l]);
  }

  // --- embedding MLP (WMMA) ---
  gemm_bt<<<dim3((E1 + 63) / 64, BATCH / 128), 256, 0, stream>>>(
      ab0, AB0LD, btE1, AB0LD, AB0LD, E1, zb1, E1LD, nullptr, 0, 1);
  gemm_bt<<<dim3((ZDIM + 63) / 64, BATCH / 128), 256, 0, stream>>>(
      zb1, E1LD, btE2, E1LD, E1LD, ZDIM, zb, ZLD, nullptr, 0, 1);

  // --- hypernetwork layers ---
  const __bf16* hcur = hb0;
  __bf16* hnext[3] = {hb1, hb2, nullptr};
  for (int l = 0; l < 3; ++l) {
    int fo = FO[l];
    gemm_bt<<<dim3((fo + 63) / 64, BATCH / 128), 256, 0, stream>>>(
        zb, ZLD, bwt[l], ZLD, ZLD, fo, nullptr, 0, bg, fo, 0);
    gemm_bt<<<dim3((fo + 63) / 64, BATCH / 128), 256, 0, stream>>>(
        zb, ZLD, swt[l], ZLD, ZLD, fo, nullptr, 0, sg, fo, 0);
    if (l == 0)        // 64-row tiles, 4 waves x 16 cols, 2 N-tiles
      hyper_layer<7, 4><<<dim3(BATCH / 64, 2), 128, 0, stream>>>(
          hcur, zb, wwt[l], fo, mv, fo);
    else if (l == 1)
      hyper_layer<4, 4><<<dim3(BATCH / 64, 2), 128, 0, stream>>>(
          hcur, zb, wwt[l], fo, mv, fo);
    else               // fo = 4: single 16-col tile per block
      hyper_layer<4, 1><<<dim3(BATCH / 16, 1), 32, 0, stream>>>(
          hcur, zb, wwt[l], fo, mv, fo);
    combine_layer<<<1024, 256, 0, stream>>>(
        mv, bg, sg, fo, hnext[l], (l == 2) ? (float*)d_out : nullptr, l < 2);
    hcur = hnext[l];
  }
}
